// EventScopeGuardHead_19344532702122
// MI455X (gfx1250) — matrix operand order repaired and checked by reading, not compiled, not run on hardware
//
#include <hip/hip_runtime.h>

typedef __attribute__((ext_vector_type(16))) _Float16 v16h;
typedef __attribute__((ext_vector_type(8)))  _Float16 v8h;
typedef __attribute__((ext_vector_type(8)))  float    v8f;

#define BSZ 4
#define NN  256
#define DD  128
#define HIDN 128

// ---------------------------------------------------------------------------
// Kernel 1: convert node latents to f16, precompute PA = h @ W1[0:128],
//           PB = h @ W1[128:256]   (one block per (b,n) row, 128 threads)
// ---------------------------------------------------------------------------
__global__ void prep_nodes(const float* __restrict__ node_latents,
                           const float* __restrict__ W1,
                           _Float16* __restrict__ hF16,
                           float* __restrict__ PA,
                           float* __restrict__ PB) {
    int row = blockIdx.x;          // b*N + n
    int h   = threadIdx.x;         // 0..127
    __shared__ float x[DD];
    float v = node_latents[(size_t)row * DD + h];
    x[h] = v;
    hF16[(size_t)row * DD + h] = (_Float16)v;
    __syncthreads();
    float pa = 0.f, pb = 0.f;
#pragma unroll 8
    for (int k = 0; k < DD; ++k) {
        float xv = x[k];
        pa = fmaf(xv, W1[(size_t)k * HIDN + h], pa);
        pb = fmaf(xv, W1[(size_t)(DD + k) * HIDN + h], pb);
    }
    PA[(size_t)row * HIDN + h] = pa;
    PB[(size_t)row * HIDN + h] = pb;
}

// ---------------------------------------------------------------------------
// Kernel 2: pack W1[256:512] (the |diff| and prod blocks) into f16 B-fragment
// layout: frag (t,n) -> [lane][16 halves]; lane = col + 16*((kk>>3)&1), half = (kk&7) + 8*(kk>>4)
// k in [0,128) -> |diff| rows (W1 row 256+k); k in [128,256) -> prod rows.
// ---------------------------------------------------------------------------
__global__ void prep_w(const float* __restrict__ W1,
                       _Float16* __restrict__ Bfrag) {
    int idx = blockIdx.x * 256 + threadIdx.x;   // 0..32767
    int k = idx >> 7;        // 0..255  (K index in pairwise GEMM)
    int h = idx & 127;       // hidden index
    int t = k >> 5, kk = k & 31;
    int n = h >> 4, cc = h & 15;
    int lane = cc + (((kk >> 3) & 1) ? 16 : 0);
    int half = (kk & 7) + ((kk >> 4) << 3);
    float w = W1[(size_t)(256 + k) * HIDN + h];
    Bfrag[(((size_t)(t * 8 + n) * 32 + lane) * 16) + half] = (_Float16)w;
}

// ---------------------------------------------------------------------------
// Kernel 3: main pairwise WMMA kernel.
// grid = (N/16 j-tiles, N/8 i-tiles, B); 256 threads = 8 waves.
// Wave w handles i = i0+w, 16 j's (one 16x128x256 f16 GEMM -> 16x128 hidden).
// ---------------------------------------------------------------------------
__global__ void __launch_bounds__(256)
pair_mlp_wmma(const _Float16* __restrict__ hF16,
              const _Float16* __restrict__ Bfrag,
              const float* __restrict__ PA,
              const float* __restrict__ PB,
              const float* __restrict__ base_logits,
              const float* __restrict__ comp_logits,
              const float* __restrict__ W1,
              const float* __restrict__ b1,
              const float* __restrict__ W2,
              const float* __restrict__ b2,
              float* __restrict__ Lraw) {
    __shared__ __align__(16) _Float16 ldsB[8 * 8 * 32 * 16]; // 64KB packed W1 frags
    __shared__ __align__(16) _Float16 ldsHI[8 * DD];         // 2KB  i-rows
    __shared__ __align__(16) _Float16 ldsHJ[16 * DD];        // 4KB  j-rows

    const int tid  = threadIdx.x;
    const int wave = tid >> 5;
    const int lane = tid & 31;
    const int c    = lane & 15;      // A row (=jj) / output column within tile
    const int sel  = lane >> 4;      // 0 or 1
    const int b    = blockIdx.z;
    const int i0   = blockIdx.y * 8;
    const int j0   = blockIdx.x * 16;
    const int i    = i0 + wave;

    // hint: keep the shared packed-weight region hot
    __builtin_prefetch(Bfrag, 0, 3);

    // ---- stage B fragments + latent tiles into LDS ----
    {
        const uint4* src = (const uint4*)Bfrag;          // 4096 x 16B
        uint4* dst = (uint4*)ldsB;
#pragma unroll
        for (int q = 0; q < 16; ++q) dst[tid + 256 * q] = src[tid + 256 * q];

        const uint4* srcJ = (const uint4*)(hF16 + ((size_t)b * NN + j0) * DD);
        ((uint4*)ldsHJ)[tid] = srcJ[tid];                // 256 x 16B
        if (tid < 128) {
            const uint4* srcI = (const uint4*)(hF16 + ((size_t)b * NN + i0) * DD);
            ((uint4*)ldsHI)[tid] = srcI[tid];            // 128 x 16B
        }
    }
    __syncthreads();

    // ---- accumulators: 8 N-tiles of 16x16 f32 ----
    v8f acc[8];
#pragma unroll
    for (int n = 0; n < 8; ++n)
#pragma unroll
        for (int e = 0; e < 8; ++e) acc[n][e] = 0.0f;

    const int kb = 8 * sel;  // per-lane K base within a 32-wide K tile

    // ---- K loop: t covers diff-K tile t and prod-K tile t+4 (same h values) ----
#pragma unroll
    for (int t = 0; t < 4; ++t) {
        int k0 = t * 32;
        v8h hiA = *(const v8h*)&ldsHI[wave * DD + k0 + kb];
        v8h hiB = *(const v8h*)&ldsHI[wave * DD + k0 + 16 + kb];
        v8h hjA = *(const v8h*)&ldsHJ[c * DD + k0 + kb];
        v8h hjB = *(const v8h*)&ldsHJ[c * DD + k0 + 16 + kb];

        v16h Ad, Ap;   // A fragments built directly in ISA layout, in registers
#pragma unroll
        for (int e = 0; e < 8; ++e) {
            float xi = (float)hiA[e], xj = (float)hjA[e];
            Ad[e]     = (_Float16)fabsf(xi - xj);
            Ap[e]     = (_Float16)(xi * xj);
            float yi = (float)hiB[e], yj = (float)hjB[e];
            Ad[e + 8] = (_Float16)fabsf(yi - yj);
            Ap[e + 8] = (_Float16)(yi * yj);
        }
#pragma unroll
        for (int n = 0; n < 8; ++n) {
            v16h Bd = *(const v16h*)&ldsB[((t * 8 + n) * 32 + lane) * 16];
            acc[n] = __builtin_amdgcn_wmma_f32_16x16x32_f16(
                false, Ad, false, Bd, (short)0, acc[n], false, false);
            v16h Bp = *(const v16h*)&ldsB[(((t + 4) * 8 + n) * 32 + lane) * 16];
            acc[n] = __builtin_amdgcn_wmma_f32_16x16x32_f16(
                false, Ap, false, Bp, (short)0, acc[n], false, false);
        }
    }

    // ---- epilogue: add PA/PB/bias/scalar-feature terms, ReLU, dot with W2 ----
    float bclip[8], bsig[8], cclip[8], csig[8];
#pragma unroll
    for (int v = 0; v < 8; ++v) {
        int M = v + 8 * sel;               // output row = jj
        int j = j0 + M;
        float bb = base_logits[((size_t)b * NN + i) * NN + j];
        bb = fminf(fmaxf(bb, -20.f), 20.f);
        float cx = comp_logits[((size_t)b * NN + i) * NN + j];
        cx = fminf(fmaxf(cx, -20.f), 20.f);
        bclip[v] = bb; bsig[v] = 1.f / (1.f + expf(-bb));
        cclip[v] = cx; csig[v] = 1.f / (1.f + expf(-cx));
    }

    float rowdot[8];
#pragma unroll
    for (int v = 0; v < 8; ++v) rowdot[v] = 0.f;

#pragma unroll
    for (int n = 0; n < 8; ++n) {
        int h = n * 16 + c;
        float pah  = PA[((size_t)b * NN + i) * HIDN + h];
        float b1h  = b1[h];
        float wbh  = W1[(size_t)512 * HIDN + h];
        float wsbh = W1[(size_t)513 * HIDN + h];
        float wch  = W1[(size_t)514 * HIDN + h];
        float wsch = W1[(size_t)515 * HIDN + h];
        float w2h  = W2[h];
#pragma unroll
        for (int v = 0; v < 8; ++v) {
            int M = v + 8 * sel;
            int j = j0 + M;
            float val = acc[n][v] + pah + b1h
                      + PB[((size_t)b * NN + j) * HIDN + h]
                      + bclip[v] * wbh + bsig[v] * wsbh
                      + cclip[v] * wch + csig[v] * wsch;
            val = fmaxf(val, 0.f);
            rowdot[v] = fmaf(val, w2h, rowdot[v]);
        }
    }

    // reduce over the 16 columns (stays within each 16-lane half, wave32)
#pragma unroll
    for (int m = 1; m < 16; m <<= 1)
#pragma unroll
        for (int v = 0; v < 8; ++v)
            rowdot[v] += __shfl_xor(rowdot[v], m, 32);

    if (c == 0) {
        float bias2 = b2[0];
#pragma unroll
        for (int v = 0; v < 8; ++v) {
            int M = v + 8 * sel;
            Lraw[((size_t)b * NN + i) * NN + (j0 + M)] = rowdot[v] + bias2;
        }
    }
}

// ---------------------------------------------------------------------------
// Kernel 4: symmetrize + diagonal mask
// ---------------------------------------------------------------------------
__global__ void symmetrize(const float* __restrict__ Lraw,
                           float* __restrict__ out) {
    int idx = blockIdx.x * 256 + threadIdx.x;   // b*N*N flat
    int b   = idx >> 16;
    int rem = idx & 65535;
    int ii  = rem >> 8, jj = rem & 255;
    float v = 0.5f * (Lraw[idx] + Lraw[((size_t)b << 16) + (jj << 8) + ii]);
    out[idx] = (ii == jj) ? -1e9f : v;
}

// ---------------------------------------------------------------------------
extern "C" void kernel_launch(void* const* d_in, const int* in_sizes, int n_in,
                              void* d_out, int out_size, void* d_ws, size_t ws_size,
                              hipStream_t stream) {
    const float* node_latents = (const float*)d_in[0];
    const float* base_logits  = (const float*)d_in[1];
    const float* comp_logits  = (const float*)d_in[2];
    const float* W1           = (const float*)d_in[3];
    const float* b1           = (const float*)d_in[4];
    const float* W2           = (const float*)d_in[5];
    const float* b2           = (const float*)d_in[6];
    float* out = (float*)d_out;

    char* ws = (char*)d_ws;
    _Float16* hF16  = (_Float16*)(ws + 0);                 // 256 KB
    _Float16* Bfrag = (_Float16*)(ws + (256 << 10));       // 64 KB
    float*    PA    = (float*)   (ws + (320 << 10));       // 512 KB
    float*    PB    = (float*)   (ws + (832 << 10));       // 512 KB
    float*    Lraw  = (float*)   (ws + (1344 << 10));      // 1 MB

    (void)in_sizes; (void)n_in; (void)out_size; (void)ws_size;

    prep_nodes<<<BSZ * NN, DD, 0, stream>>>(node_latents, W1, hF16, PA, PB);
    prep_w<<<128, 256, 0, stream>>>(W1, Bfrag);
    pair_mlp_wmma<<<dim3(NN / 16, NN / 8, BSZ), 256, 0, stream>>>(
        hF16, Bfrag, PA, PB, base_logits, comp_logits, W1, b1, W2, b2, Lraw);
    symmetrize<<<(BSZ * NN * NN) / 256, 256, 0, stream>>>(Lraw, out);
}
